// MoE_4088808866374
// MI455X (gfx1250) — matrix operand order repaired and checked by reading, not compiled, not run on hardware
//
#include <hip/hip_runtime.h>
#include <math.h>

// ---------------- problem constants ----------------
#define D_DIM 1024     // model dim
#define F_DIM 2816     // ffn dim
#define E_NUM 8        // experts
#define A_TOK 4096     // B*S tokens
#define MT    32       // M tile (tokens per block)
#define KT    32       // K step (one bf16 WMMA 16x16x32)
#define NT    128      // N tile per block (8 waves x 16)
#define KP    40       // padded LDS K-stride (halfwords): 80B rows, conflict-free, 16B aligned

typedef __bf16 bf16_t;
typedef __attribute__((ext_vector_type(16))) __bf16 v16bf;
typedef __attribute__((ext_vector_type(8)))  __bf16 v8bf;
typedef __attribute__((ext_vector_type(8)))  float  v8f;

union V16 { v16bf v; v8bf h[2]; };

__device__ __forceinline__ unsigned pack_bf16(float a, float b) {
    union { bf16_t h[2]; unsigned u; } p;
    p.h[0] = (bf16_t)a; p.h[1] = (bf16_t)b;
    return p.u;
}
__device__ __forceinline__ float fast_sigmoid(float x) {
    return __builtin_amdgcn_rcpf(1.0f + __expf(-x));
}
// CDNA5 async DMA: global -> LDS, tracked by ASYNCcnt.
__device__ __forceinline__ void async_ld_b128(unsigned lds_off, const void* gaddr) {
    asm volatile("global_load_async_to_lds_b128 %0, %1, off"
                 :: "v"(lds_off), "v"(gaddr) : "memory");
}
__device__ __forceinline__ void async_ld_b64(unsigned lds_off, const void* gaddr) {
    asm volatile("global_load_async_to_lds_b64 %0, %1, off"
                 :: "v"(lds_off), "v"(gaddr) : "memory");
}
__device__ __forceinline__ void wait_asynccnt0() {
    asm volatile("s_wait_asynccnt 0x0" ::: "memory");
}
__device__ __forceinline__ unsigned lds_addr_of(const void* p) {
    return (unsigned)(unsigned long long)p;   // flat shared addr: low 32 bits = LDS offset
}

// ---------------- setup kernels ----------------
__global__ void moe_init(int* counts, int* cur) {
    if (threadIdx.x < E_NUM) { counts[threadIdx.x] = 0; cur[threadIdx.x] = 0; }
}

__global__ __launch_bounds__(256) void moe_router(
    const float* __restrict__ x, const float* __restrict__ router,
    float* __restrict__ gate, int* __restrict__ eid, int* __restrict__ counts)
{
    int t    = blockIdx.x;
    int w    = threadIdx.x >> 5;
    int lane = threadIdx.x & 31;
    const float* xr = x + (size_t)t * D_DIM;
    float s = 0.f;
    for (int k = lane; k < D_DIM; k += 32)
        s += xr[k] * router[(size_t)k * E_NUM + w];
    for (int off = 16; off; off >>= 1) s += __shfl_xor(s, off, 32);
    __shared__ float sc[E_NUM];
    if (lane == 0) sc[w] = s;
    __syncthreads();
    if (threadIdx.x == 0) {
        int be = 0; float bv = sc[0];
        for (int e = 1; e < E_NUM; ++e) if (sc[e] > bv) { bv = sc[e]; be = e; }
        gate[t] = fast_sigmoid(bv);
        eid[t]  = be;
        atomicAdd(&counts[be], 1);
    }
}

__global__ void moe_scan(const int* counts, int* base, int* cur) {
    if (threadIdx.x == 0) {
        int acc = 0;
        for (int e = 0; e < E_NUM; ++e) { base[e] = acc; cur[e] = acc; acc += counts[e]; }
    }
}

__global__ void moe_scatter(const int* __restrict__ eid, int* cur, int* __restrict__ perm) {
    int t = blockIdx.x * blockDim.x + threadIdx.x;
    if (t < A_TOK) {
        int pos = atomicAdd(&cur[eid[t]], 1);
        perm[pos] = t;
    }
}

// transpose+convert: src [K][N] fp32 -> dst [N][K] bf16 (per blockIdx.z matrix)
__global__ __launch_bounds__(256) void moe_conv_t(
    const float* __restrict__ src, bf16_t* __restrict__ dst, int K, int N)
{
    __shared__ float t[32][33];
    size_t off = (size_t)blockIdx.z * K * N;
    src += off; dst += off;
    int k0 = blockIdx.y * 32, n0 = blockIdx.x * 32;
    int tx = threadIdx.x & 31, ty = threadIdx.x >> 5;
#pragma unroll
    for (int i = 0; i < 4; ++i)
        t[ty + 8 * i][tx] = src[(size_t)(k0 + ty + 8 * i) * N + n0 + tx];
    __syncthreads();
#pragma unroll
    for (int i = 0; i < 4; ++i)
        dst[(size_t)(n0 + ty + 8 * i) * K + k0 + tx] = (bf16_t)t[tx][ty + 8 * i];
}

// activations -> bf16: xs[t][d] = x[t][d]; xg[slot][d] = x[perm[slot]][d] * gate
__global__ __launch_bounds__(256) void moe_xconv(
    const float* __restrict__ x, const float* __restrict__ gate,
    const int* __restrict__ perm, bf16_t* __restrict__ xs, bf16_t* __restrict__ xg)
{
    int g = blockIdx.x;
    int tid = threadIdx.x;
    const float* xr = x + (size_t)g * D_DIM;
    unsigned* od = (unsigned*)(xs + (size_t)g * D_DIM);
#pragma unroll
    for (int i = 0; i < (D_DIM / 2) / 256; ++i) {
        int j = tid + i * 256;
        float2 v = *(const float2*)(xr + 2 * j);
        od[j] = pack_bf16(v.x, v.y);
    }
    int tok = perm[g];
    float s = gate[tok];
    const float* xr2 = x + (size_t)tok * D_DIM;
    unsigned* od2 = (unsigned*)(xg + (size_t)g * D_DIM);
#pragma unroll
    for (int i = 0; i < (D_DIM / 2) / 256; ++i) {
        int j = tid + i * 256;
        float2 v = *(const float2*)(xr2 + 2 * j);
        od2[j] = pack_bf16(v.x * s, v.y * s);
    }
}

// ---------------- GEMM 1: mid = silu(XA*W1t) .* (XA*W3t) ----------------
__global__ __launch_bounds__(256) void moe_gemm1(
    const bf16_t* __restrict__ XA,
    const bf16_t* __restrict__ W1all, const bf16_t* __restrict__ W3all,
    const int* __restrict__ counts, const int* __restrict__ base,
    bf16_t* __restrict__ mid, int mode)
{
    __shared__ __align__(16) bf16_t As [2][MT * KP];
    __shared__ __align__(16) bf16_t B1s[2][NT * KP];
    __shared__ __align__(16) bf16_t B3s[2][NT * KP];

    int tid = threadIdx.x;
    const bf16_t *W1, *W3;
    int rowBase, rowsValid;
    if (mode == 0) {
        W1 = W1all; W3 = W3all;
        rowBase = blockIdx.y * MT; rowsValid = MT;
    } else {
        int e  = blockIdx.y >> 7;
        int mt = blockIdx.y & 127;
        int cnt = counts[e];
        if (mt * MT >= cnt) return;
        rowBase   = base[e] + mt * MT;
        rowsValid = cnt - mt * MT; if (rowsValid > MT) rowsValid = MT;
        W1 = W1all + (size_t)e * D_DIM * F_DIM;
        W3 = W3all + (size_t)e * D_DIM * F_DIM;
    }
    int n0 = blockIdx.x * NT;

    int w = tid >> 5, lane = tid & 31;
    int nrow = w * 16 + (lane & 15);
    int kbB  = (lane >> 4) * 8;
    int rA   = lane & 15;
    int kbA  = (lane >> 4) * 8;

    // uniform async lane assignments: A = 32 rows x 8 B64-chunks (all 256 lanes)
    int ar = tid >> 3, ac = tid & 7;
    int arow = rowBase + ar; if (arow >= A_TOK) arow = A_TOK - 1;   // tail clamp
    const bf16_t* aSrc = XA + (size_t)arow * D_DIM + 4 * ac;
    int bn = tid >> 2, bc = tid & 3;                 // B = 128 rows x 4 chunks x 2 halves
    const bf16_t* b1p0 = W1 + (size_t)(n0 + bn)      * D_DIM + 8 * bc;
    const bf16_t* b1p1 = W1 + (size_t)(n0 + bn + 64) * D_DIM + 8 * bc;
    const bf16_t* b3p0 = W3 + (size_t)(n0 + bn)      * D_DIM + 8 * bc;
    const bf16_t* b3p1 = W3 + (size_t)(n0 + bn + 64) * D_DIM + 8 * bc;
    unsigned ldsA   = lds_addr_of(&As [0][ar * KP + 4 * ac]);
    unsigned ldsB10 = lds_addr_of(&B1s[0][ bn       * KP + 8 * bc]);
    unsigned ldsB11 = lds_addr_of(&B1s[0][(bn + 64) * KP + 8 * bc]);
    unsigned ldsB30 = lds_addr_of(&B3s[0][ bn       * KP + 8 * bc]);
    unsigned ldsB31 = lds_addr_of(&B3s[0][(bn + 64) * KP + 8 * bc]);
    const unsigned aStride = MT * KP * 2, bStride = NT * KP * 2;

    auto stage = [&](int buf) {   // buf is a literal at every call site
        async_ld_b64 (ldsA   + buf * aStride, aSrc);
        async_ld_b128(ldsB10 + buf * bStride, b1p0);
        async_ld_b128(ldsB11 + buf * bStride, b1p1);
        async_ld_b128(ldsB30 + buf * bStride, b3p0);
        async_ld_b128(ldsB31 + buf * bStride, b3p1);
        aSrc += KT; b1p0 += KT; b1p1 += KT; b3p0 += KT; b3p1 += KT;
    };

    v8f c1_0 = {}, c1_1 = {}, c3_0 = {}, c3_1 = {};

    auto step = [&](int buf, bool doStage) {
        V16 a0, a1, b1, b3;
        a0.h[0] = *(const v8bf*)&As [buf][ rA       * KP + kbA     ];
        a0.h[1] = *(const v8bf*)&As [buf][ rA       * KP + kbA + 16];
        a1.h[0] = *(const v8bf*)&As [buf][(rA + 16) * KP + kbA     ];
        a1.h[1] = *(const v8bf*)&As [buf][(rA + 16) * KP + kbA + 16];
        b1.h[0] = *(const v8bf*)&B1s[buf][ nrow     * KP + kbB     ];
        b1.h[1] = *(const v8bf*)&B1s[buf][ nrow     * KP + kbB + 16];
        b3.h[0] = *(const v8bf*)&B3s[buf][ nrow     * KP + kbB     ];
        b3.h[1] = *(const v8bf*)&B3s[buf][ nrow     * KP + kbB + 16];
        if (doStage) stage(buf ^ 1);           // async DMA overlaps the WMMAs below
        c1_0 = __builtin_amdgcn_wmma_f32_16x16x32_bf16(false, a0.v, false, b1.v, (short)0, c1_0, false, false);
        c1_1 = __builtin_amdgcn_wmma_f32_16x16x32_bf16(false, a1.v, false, b1.v, (short)0, c1_1, false, false);
        c3_0 = __builtin_amdgcn_wmma_f32_16x16x32_bf16(false, a0.v, false, b3.v, (short)0, c3_0, false, false);
        c3_1 = __builtin_amdgcn_wmma_f32_16x16x32_bf16(false, a1.v, false, b3.v, (short)0, c3_1, false, false);
        wait_asynccnt0();
        __syncthreads();
    };

    const int NSTEP = D_DIM / KT;   // 32 (even)
    stage(0);
    wait_asynccnt0();
    __syncthreads();
    for (int s = 0; s + 2 < NSTEP; s += 2) {   // static buffer parity, no selects
        step(0, true);
        step(1, true);
    }
    step(0, true);
    step(1, false);

    int nglob = n0 + nrow;
    int mhi   = (lane >> 4) * 8;
    bf16_t* midp = mid + nglob;
    if (rowsValid == MT) {
#pragma unroll
        for (int v = 0; v < 8; ++v) {
            int m = mhi + v;
            midp[(size_t)(rowBase + m)      * F_DIM] = (bf16_t)(c1_0[v] * fast_sigmoid(c1_0[v]) * c3_0[v]);
            midp[(size_t)(rowBase + 16 + m) * F_DIM] = (bf16_t)(c1_1[v] * fast_sigmoid(c1_1[v]) * c3_1[v]);
        }
    } else {
#pragma unroll
        for (int v = 0; v < 8; ++v) {
            int m = mhi + v;
            if (m < rowsValid)
                midp[(size_t)(rowBase + m)      * F_DIM] = (bf16_t)(c1_0[v] * fast_sigmoid(c1_0[v]) * c3_0[v]);
            if (16 + m < rowsValid)
                midp[(size_t)(rowBase + 16 + m) * F_DIM] = (bf16_t)(c1_1[v] * fast_sigmoid(c1_1[v]) * c3_1[v]);
        }
    }
}

// ---------------- GEMM 2: out(+)= mid * W2t ----------------
__global__ __launch_bounds__(256) void moe_gemm2(
    const bf16_t* __restrict__ mid, const bf16_t* __restrict__ W2all,
    const int* __restrict__ perm, const int* __restrict__ counts, const int* __restrict__ base,
    float* __restrict__ out, int mode)
{
    __shared__ __align__(16) bf16_t As[2][MT * KP];
    __shared__ __align__(16) bf16_t Bs[2][NT * KP];
    __shared__ int rowTok[MT];

    int tid = threadIdx.x;
    const bf16_t* W2;
    int rowBase, rowsValid;
    if (mode == 0) {
        W2 = W2all;
        rowBase = blockIdx.y * MT; rowsValid = MT;
    } else {
        int e  = blockIdx.y >> 7;
        int mt = blockIdx.y & 127;
        int cnt = counts[e];
        if (mt * MT >= cnt) return;
        rowBase   = base[e] + mt * MT;
        rowsValid = cnt - mt * MT; if (rowsValid > MT) rowsValid = MT;
        W2 = W2all + (size_t)e * F_DIM * D_DIM;
    }
    int n0 = blockIdx.x * NT;

    if (tid < MT) {
        int g = rowBase + tid;
        rowTok[tid] = (tid < rowsValid) ? ((mode == 0) ? g : perm[g]) : 0;
    }

    int w = tid >> 5, lane = tid & 31;
    int nrow = w * 16 + (lane & 15);
    int kbB  = (lane >> 4) * 8;
    int rA   = lane & 15;
    int kbA  = (lane >> 4) * 8;

    int ar = tid >> 3, ac = tid & 7;
    int arow = rowBase + ar; if (arow >= A_TOK) arow = A_TOK - 1;
    const bf16_t* aSrc = mid + (size_t)arow * F_DIM + 4 * ac;
    int bn = tid >> 2, bc = tid & 3;
    const bf16_t* bp0 = W2 + (size_t)(n0 + bn)      * F_DIM + 8 * bc;
    const bf16_t* bp1 = W2 + (size_t)(n0 + bn + 64) * F_DIM + 8 * bc;
    unsigned ldsA  = lds_addr_of(&As[0][ar * KP + 4 * ac]);
    unsigned ldsB0 = lds_addr_of(&Bs[0][ bn       * KP + 8 * bc]);
    unsigned ldsB1 = lds_addr_of(&Bs[0][(bn + 64) * KP + 8 * bc]);
    const unsigned aStride = MT * KP * 2, bStride = NT * KP * 2;

    auto stage = [&](int buf) {
        async_ld_b64 (ldsA  + buf * aStride, aSrc);
        async_ld_b128(ldsB0 + buf * bStride, bp0);
        async_ld_b128(ldsB1 + buf * bStride, bp1);
        aSrc += KT; bp0 += KT; bp1 += KT;
    };

    v8f c_0 = {}, c_1 = {};

    auto step = [&](int buf, bool doStage) {
        V16 a0, a1, b;
        a0.h[0] = *(const v8bf*)&As[buf][ rA       * KP + kbA     ];
        a0.h[1] = *(const v8bf*)&As[buf][ rA       * KP + kbA + 16];
        a1.h[0] = *(const v8bf*)&As[buf][(rA + 16) * KP + kbA     ];
        a1.h[1] = *(const v8bf*)&As[buf][(rA + 16) * KP + kbA + 16];
        b.h[0]  = *(const v8bf*)&Bs[buf][ nrow     * KP + kbB     ];
        b.h[1]  = *(const v8bf*)&Bs[buf][ nrow     * KP + kbB + 16];
        if (doStage) stage(buf ^ 1);
        c_0 = __builtin_amdgcn_wmma_f32_16x16x32_bf16(false, a0.v, false, b.v, (short)0, c_0, false, false);
        c_1 = __builtin_amdgcn_wmma_f32_16x16x32_bf16(false, a1.v, false, b.v, (short)0, c_1, false, false);
        wait_asynccnt0();
        __syncthreads();
    };

    const int NSTEP = F_DIM / KT;   // 88 (even)
    stage(0);
    wait_asynccnt0();
    __syncthreads();
    for (int s = 0; s + 2 < NSTEP; s += 2) {
        step(0, true);
        step(1, true);
    }
    step(0, true);
    step(1, false);

    int d   = n0 + nrow;
    int mhi = (lane >> 4) * 8;
    if (rowsValid == MT) {
#pragma unroll
        for (int v = 0; v < 8; ++v) {
            int m = mhi + v;
            size_t o0 = (size_t)rowTok[m]      * D_DIM + d;
            size_t o1 = (size_t)rowTok[16 + m] * D_DIM + d;
            if (mode == 0) { out[o0] = c_0[v];  out[o1] = c_1[v]; }
            else           { out[o0] += c_0[v]; out[o1] += c_1[v]; }
        }
    } else {
#pragma unroll
        for (int v = 0; v < 8; ++v) {
            int m = mhi + v;
            if (m < rowsValid) {
                size_t o = (size_t)rowTok[m] * D_DIM + d;
                if (mode == 0) out[o] = c_0[v]; else out[o] += c_0[v];
            }
            if (16 + m < rowsValid) {
                size_t o = (size_t)rowTok[16 + m] * D_DIM + d;
                if (mode == 0) out[o] = c_1[v]; else out[o] += c_1[v];
            }
        }
    }
}

// ---------------- host glue ----------------
extern "C" void kernel_launch(void* const* d_in, const int* in_sizes, int n_in,
                              void* d_out, int out_size, void* d_ws, size_t ws_size,
                              hipStream_t stream)
{
    const float* x      = (const float*)d_in[0];
    const float* router = (const float*)d_in[1];
    const float* w1     = (const float*)d_in[2];
    const float* w3     = (const float*)d_in[3];
    const float* w2     = (const float*)d_in[4];
    const float* ws1    = (const float*)d_in[5];
    const float* ws3    = (const float*)d_in[6];
    const float* ws2    = (const float*)d_in[7];
    float* out = (float*)d_out;

    char* ws = (char*)d_ws;
    const size_t MB = 1ull << 20;
    float*  gate   = (float*)(ws + 0);
    int*    eid    = (int*)  (ws + 16384);
    int*    counts = (int*)  (ws + 32768);
    int*    base   = (int*)  (ws + 32768 + 64);
    int*    cur    = (int*)  (ws + 32768 + 128);
    int*    perm   = (int*)  (ws + 65536);
    bf16_t* xs   = (bf16_t*)(ws + 1   * MB);
    bf16_t* xg   = (bf16_t*)(ws + 10  * MB);
    bf16_t* mid  = (bf16_t*)(ws + 20  * MB);
    bf16_t* w1t  = (bf16_t*)(ws + 44  * MB);
    bf16_t* w3t  = (bf16_t*)(ws + 92  * MB);
    bf16_t* w2t  = (bf16_t*)(ws + 140 * MB);
    bf16_t* ws1t = (bf16_t*)(ws + 188 * MB);
    bf16_t* ws3t = (bf16_t*)(ws + 194 * MB);
    bf16_t* ws2t = (bf16_t*)(ws + 200 * MB);

    moe_init   <<<1, 64, 0, stream>>>(counts, cur);
    moe_router <<<A_TOK, 256, 0, stream>>>(x, router, gate, eid, counts);
    moe_scan   <<<1, 1, 0, stream>>>(counts, base, cur);
    moe_scatter<<<A_TOK / 256, 256, 0, stream>>>(eid, cur, perm);
    moe_xconv  <<<A_TOK, 256, 0, stream>>>(x, gate, perm, xs, xg);

    moe_conv_t<<<dim3(F_DIM / 32, D_DIM / 32, E_NUM), 256, 0, stream>>>(w1,  w1t,  D_DIM, F_DIM);
    moe_conv_t<<<dim3(F_DIM / 32, D_DIM / 32, E_NUM), 256, 0, stream>>>(w3,  w3t,  D_DIM, F_DIM);
    moe_conv_t<<<dim3(D_DIM / 32, F_DIM / 32, E_NUM), 256, 0, stream>>>(w2,  w2t,  F_DIM, D_DIM);
    moe_conv_t<<<dim3(F_DIM / 32, D_DIM / 32, 1),     256, 0, stream>>>(ws1, ws1t, D_DIM, F_DIM);
    moe_conv_t<<<dim3(F_DIM / 32, D_DIM / 32, 1),     256, 0, stream>>>(ws3, ws3t, D_DIM, F_DIM);
    moe_conv_t<<<dim3(D_DIM / 32, F_DIM / 32, 1),     256, 0, stream>>>(ws2, ws2t, F_DIM, D_DIM);

    dim3 g1s(F_DIM / NT, A_TOK / MT);
    moe_gemm1<<<g1s, 256, 0, stream>>>(xs, ws1t, ws3t, counts, base, mid, 0);
    dim3 g2s(D_DIM / NT, A_TOK / MT);
    moe_gemm2<<<g2s, 256, 0, stream>>>(mid, ws2t, perm, counts, base, out, 0);

    dim3 g1r(F_DIM / NT, E_NUM * (A_TOK / MT));
    moe_gemm1<<<g1r, 256, 0, stream>>>(xg, w1t, w3t, counts, base, mid, 1);
    dim3 g2r(D_DIM / NT, E_NUM * (A_TOK / MT));
    moe_gemm2<<<g2r, 256, 0, stream>>>(mid, w2t, perm, counts, base, out, 1);
}
